// CausalSelfAttention_31559419691190
// MI455X (gfx1250) — matrix operand order repaired and checked by reading, not compiled, not run on hardware
//
#include <hip/hip_runtime.h>

typedef __attribute__((ext_vector_type(16))) __bf16          v16bf;
typedef __attribute__((ext_vector_type(8)))  __bf16          v8bf;
typedef __attribute__((ext_vector_type(8)))  float           v8f;
typedef __attribute__((ext_vector_type(8)))  unsigned short  v8u;
typedef __attribute__((ext_vector_type(4)))  unsigned short  v4u;

#define DEV __device__ __forceinline__

constexpr int Bc  = 2;
constexpr int Sc  = 2048;
constexpr int Dc  = 2048;
constexpr int Hc  = 16;
constexpr int DKc = 128;
constexpr int N3c = 3 * Dc;     // 6144
constexpr int Mc  = Bc * Sc;    // 4096

// ---------------- helpers ----------------

DEV unsigned short f2bf(float f) {
  unsigned u = __float_as_uint(f);
  u += 0x7FFFu + ((u >> 16) & 1u);          // round-to-nearest-even
  return (unsigned short)(u >> 16);
}

DEV v16bf cat16(v8bf lo, v8bf hi) {
  return __builtin_shufflevector(lo, hi, 0,1,2,3,4,5,6,7,8,9,10,11,12,13,14,15);
}

DEV v8f wmma_bf16(v16bf a, v16bf b, v8f c) {
  // (neg_a, A, neg_b, B, c_mod, C, reuse_a, reuse_b)
  return __builtin_amdgcn_wmma_f32_16x16x32_bf16(false, a, false, b, (short)0, c, false, false);
}

DEV float xormax16(float v) {
  v = fmaxf(v, __shfl_xor(v, 1, 32));
  v = fmaxf(v, __shfl_xor(v, 2, 32));
  v = fmaxf(v, __shfl_xor(v, 4, 32));
  v = fmaxf(v, __shfl_xor(v, 8, 32));
  return v;
}
DEV float xorsum16(float v) {
  v += __shfl_xor(v, 1, 32);
  v += __shfl_xor(v, 2, 32);
  v += __shfl_xor(v, 4, 32);
  v += __shfl_xor(v, 8, 32);
  return v;
}

// LDS byte address of a __shared__ object: low 32 bits of the flat address
DEV unsigned ldsAddr(const void* p) { return (unsigned)(unsigned long long)(uintptr_t)p; }

// CDNA5 async DMA: global -> LDS, 16 bytes per lane, tracked by ASYNCcnt.
DEV void asyncLoadB128(unsigned lds, const void* g) {
  unsigned long long ga = (unsigned long long)(uintptr_t)g;
  asm volatile("global_load_async_to_lds_b128 %0, %1, off"
               :: "v"(lds), "v"(ga) : "memory");
}
DEV void waitAsync8()  { asm volatile("s_wait_asynccnt 8" ::: "memory"); }
DEV void waitAsync0()  { asm volatile("s_wait_asynccnt 0" ::: "memory"); }

// ---------------- conversion kernels ----------------

__global__ __launch_bounds__(256) void cvt_f32_bf16_kernel(
    const float* __restrict__ src, unsigned short* __restrict__ dst, int n) {
  int i = (blockIdx.x * 256 + threadIdx.x) * 4;
  if (i + 3 < n) {
    const float4 v = *(const float4*)(src + i);
    v4u o;
    o.x = f2bf(v.x); o.y = f2bf(v.y); o.z = f2bf(v.z); o.w = f2bf(v.w);
    *(v4u*)(dst + i) = o;
  }
}

// src: (K x N) fp32 row-major  ->  dst: (N x K) bf16 row-major (k contiguous)
__global__ __launch_bounds__(256) void transpose_f32_bf16_kernel(
    const float* __restrict__ src, unsigned short* __restrict__ dst, int K, int N) {
  __shared__ float tile[32][33];
  const int n0 = blockIdx.x * 32;
  const int k0 = blockIdx.y * 32;
  const int tx = threadIdx.x & 31;
  const int ty = threadIdx.x >> 5;          // 0..7
#pragma unroll
  for (int r = 0; r < 4; ++r)
    tile[ty + r * 8][tx] = src[(size_t)(k0 + ty + r * 8) * N + n0 + tx];
  __syncthreads();
#pragma unroll
  for (int r = 0; r < 4; ++r)
    dst[(size_t)(n0 + ty + r * 8) * K + k0 + tx] = f2bf(tile[tx][ty + r * 8]);
}

// ---------------- shared GEMM mainloop (double-buffered, async LDS fill) ----
// C(128x128 per block) = A(M x kdim, bf16 row-major) * Bt^T  (Bt: N x kdim bf16)
// 8 waves: waveM = w&3 (32 rows each), waveN = w>>2 (64 cols each)

DEV void gemm_tile(const unsigned short* __restrict__ A,
                   const unsigned short* __restrict__ Bt,
                   int m0, int n0, int kdim,
                   unsigned short (*sA)[128 * 64], unsigned short (*sB)[128 * 64],
                   v8f acc[2][4]) {
  const int tid   = threadIdx.x;
  const int lane  = tid & 31;
  const int w     = tid >> 5;
  const int waveM = w & 3;
  const int waveN = w >> 2;
  const int half  = lane >> 4;
  const int l15   = lane & 15;

  const int ldRow = tid >> 1;               // 0..127
  const int ldOff = (tid & 1) * 32;         // element offset: 0 or 32 (64B each)
  const unsigned short* aBase = A  + (size_t)(m0 + ldRow) * kdim + ldOff;
  const unsigned short* bBase = Bt + (size_t)(n0 + ldRow) * kdim + ldOff;
  unsigned aDst[2], bDst[2];
  aDst[0] = ldsAddr(&sA[0][ldRow * 64 + ldOff]);
  aDst[1] = ldsAddr(&sA[1][ldRow * 64 + ldOff]);
  bDst[0] = ldsAddr(&sB[0][ldRow * 64 + ldOff]);
  bDst[1] = ldsAddr(&sB[1][ldRow * 64 + ldOff]);

  // prologue: async-fill buffer 0 with tile k=0 (8 x b128 per lane)
#pragma unroll
  for (int j = 0; j < 4; ++j) {
    asyncLoadB128(aDst[0] + j * 16, aBase + j * 8);
    asyncLoadB128(bDst[0] + j * 16, bBase + j * 8);
  }

  int buf = 0;
  for (int kt = 0; kt < kdim; kt += 64) {
    const bool more = (kt + 64) < kdim;
    if (more) {
      // async-fill the other buffer with the next tile while we compute
#pragma unroll
      for (int j = 0; j < 4; ++j) {
        asyncLoadB128(aDst[buf ^ 1] + j * 16, aBase + kt + 64 + j * 8);
        asyncLoadB128(bDst[buf ^ 1] + j * 16, bBase + kt + 64 + j * 8);
      }
      waitAsync8();                         // our 8 loads for tile kt are done
    } else {
      waitAsync0();
    }
    __syncthreads();                        // everyone's tile-kt loads landed

    const unsigned short* cA = sA[buf];
    const unsigned short* cB = sB[buf];
#pragma unroll
    for (int ks = 0; ks < 64; ks += 32) {
      v16bf af[2], bfr[4];
      const int offA = ks + half * 8;       // A layout: {0..7,16..23} / {8..15,24..31}
#pragma unroll
      for (int mt = 0; mt < 2; ++mt) {
        const int row = waveM * 32 + mt * 16 + l15;
        af[mt] = cat16(*(const v8bf*)&cA[row * 64 + offA],
                       *(const v8bf*)&cA[row * 64 + offA + 16]);
      }
      const int offB = ks + half * 8;       // B layout: {0..7,16..23} / {8..15,24..31}
#pragma unroll
      for (int nt = 0; nt < 4; ++nt) {
        const int row = waveN * 64 + nt * 16 + l15;
        bfr[nt] = cat16(*(const v8bf*)&cB[row * 64 + offB],
                        *(const v8bf*)&cB[row * 64 + offB + 16]);
      }
#pragma unroll
      for (int mt = 0; mt < 2; ++mt)
#pragma unroll
        for (int nt = 0; nt < 4; ++nt)
          acc[mt][nt] = wmma_bf16(af[mt], bfr[nt], acc[mt][nt]);
    }
    __syncthreads();                        // all waves done reading buf
    buf ^= 1;
  }
}

// ---------------- kernel A: QKV projection ----------------

__global__ __launch_bounds__(256) void qkv_gemm_kernel(
    const unsigned short* __restrict__ Xbf, const unsigned short* __restrict__ WqkvT,
    const float* __restrict__ bqkv,
    unsigned short* __restrict__ Q, unsigned short* __restrict__ K,
    unsigned short* __restrict__ V) {
  __shared__ __attribute__((aligned(16))) unsigned short sA[2][128 * 64];
  __shared__ __attribute__((aligned(16))) unsigned short sB[2][128 * 64];
  v8f acc[2][4];
#pragma unroll
  for (int mt = 0; mt < 2; ++mt)
#pragma unroll
    for (int nt = 0; nt < 4; ++nt) acc[mt][nt] = v8f{};

  const int m0 = blockIdx.y * 128;
  const int n0 = blockIdx.x * 128;
  gemm_tile(Xbf, WqkvT, m0, n0, Dc, sA, sB, acc);

  const int lane  = threadIdx.x & 31;
  const int w     = threadIdx.x >> 5;
  const int waveM = w & 3, waveN = w >> 2;
  const int half  = lane >> 4, l15 = lane & 15;
#pragma unroll
  for (int nt = 0; nt < 4; ++nt) {
    const int colN = n0 + waveN * 64 + nt * 16 + l15;
    const float bias = bqkv[colN];
    const int sel = colN >> 11;             // 0=Q 1=K 2=V
    const int rem = colN & 2047;
    const int h = rem >> 7, d = rem & 127;
    unsigned short* dst = (sel == 0) ? Q : (sel == 1) ? K : V;
#pragma unroll
    for (int mt = 0; mt < 2; ++mt) {
#pragma unroll
      for (int i = 0; i < 8; ++i) {
        const int rowM = m0 + waveM * 32 + mt * 16 + i + 8 * half;
        const int b = rowM >> 11, s = rowM & 2047;
        dst[(((size_t)(b * Hc + h)) * Sc + s) * DKc + d] = f2bf(acc[mt][nt][i] + bias);
      }
    }
  }
}

// ---------------- kernel B: causal flash attention ----------------
// grid (S/64, H, B), 128 threads (4 waves), wave w owns q-rows [qb*64+w*16, +16)

__global__ __launch_bounds__(128) void attn_kernel(
    const unsigned short* __restrict__ Q, const unsigned short* __restrict__ K,
    const unsigned short* __restrict__ V, const unsigned char* __restrict__ seqmask,
    unsigned short* __restrict__ ctx) {
  __shared__ __attribute__((aligned(16))) unsigned short sK[32 * 128];   // [key][d]
  __shared__ __attribute__((aligned(16))) unsigned short sVt[128 * 32];  // [d][key]
  __shared__ __attribute__((aligned(16))) unsigned short sP[4][16 * 32]; // per-wave P

  const int tid  = threadIdx.x;
  const int lane = tid & 31;
  const int w    = tid >> 5;
  const int half = lane >> 4;
  const int l15  = lane & 15;
  const int qb   = blockIdx.x;
  const int h    = blockIdx.y;
  const int b    = blockIdx.z;
  const int qBase = qb * 64 + w * 16;
  const size_t headOff = ((size_t)(b * Hc + h)) * Sc * DKc;
  const unsigned short* Qh = Q + headOff;
  const unsigned short* Kh = K + headOff;
  const unsigned short* Vh = V + headOff;

  // resident Q fragments (16 rows x 128 dk = 4 A-frags)
  v16bf qf[4];
#pragma unroll
  for (int kk = 0; kk < 4; ++kk) {
    const int offA = kk * 32 + half * 8;
    const unsigned short* p = Qh + (size_t)(qBase + l15) * DKc + offA;
    qf[kk] = cat16(*(const v8bf*)p, *(const v8bf*)(p + 16));
  }

  v8f o[8];
#pragma unroll
  for (int t = 0; t < 8; ++t) o[t] = v8f{};
  float mi[8], li[8];
#pragma unroll
  for (int i = 0; i < 8; ++i) { mi[i] = -1e30f; li[i] = 0.0f; }

  const float scale = 0.08838834764831845f;   // 1/sqrt(128)
  const int kEnd = qb * 64 + 64;
  const int ldRow = tid >> 2;                 // key 0..31
  const int ldSeg = tid & 3;                  // d-segment, 32 elements each
  const unsigned kDstLds = ldsAddr(&sK[ldRow * 128 + ldSeg * 32]);

  for (int kb = 0; kb < kEnd; kb += 32) {
    __syncthreads();                          // tiles free to overwrite
    // K tile: [key][d] via async DMA straight into LDS
    const unsigned short* kSrc = Kh + (size_t)(kb + ldRow) * DKc + ldSeg * 32;
#pragma unroll
    for (int j = 0; j < 4; ++j)
      asyncLoadB128(kDstLds + j * 16, kSrc + j * 8);
    // V tile transposed: [d][key] (needs VGPR round-trip for the transpose)
    const unsigned short* vSrc = Vh + (size_t)(kb + ldRow) * DKc + ldSeg * 32;
    v8u vv[4];
#pragma unroll
    for (int j = 0; j < 4; ++j) vv[j] = *(const v8u*)(vSrc + j * 8);
#pragma unroll
    for (int j = 0; j < 4; ++j)
#pragma unroll
      for (int e = 0; e < 8; ++e)
        sVt[(ldSeg * 32 + j * 8 + e) * 32 + ldRow] = vv[j][e];
    waitAsync0();
    __syncthreads();

    // scores: S(16x32) = Q(16x128) * K^T(128x32)
    v8f sc[2];
    sc[0] = v8f{}; sc[1] = v8f{};
#pragma unroll
    for (int kk = 0; kk < 4; ++kk) {
      const int dOff = kk * 32 + half * 8;
#pragma unroll
      for (int nt = 0; nt < 2; ++nt) {
        const int key = nt * 16 + l15;
        v16bf bfr = cat16(*(const v8bf*)&sK[key * 128 + dOff],
                          *(const v8bf*)&sK[key * 128 + dOff + 16]);
        sc[nt] = wmma_bf16(qf[kk], bfr, sc[nt]);
      }
    }

    // scale + causal + sequence mask
    float pv[2][8];
    const int col0 = kb + l15;
    const bool sm0 = seqmask[(size_t)b * Sc + col0] != 0;
    const bool sm1 = seqmask[(size_t)b * Sc + col0 + 16] != 0;
#pragma unroll
    for (int nt = 0; nt < 2; ++nt) {
#pragma unroll
      for (int i = 0; i < 8; ++i) {
        const int col = col0 + nt * 16;
        const int row = qBase + i + 8 * half;
        const float v = sc[nt][i] * scale;
        const bool ok = (col <= row) && (nt ? sm1 : sm0);
        pv[nt][i] = ok ? v : -1e30f;
      }
    }

    // online softmax (row reductions across the 16 lanes of each half)
#pragma unroll
    for (int i = 0; i < 8; ++i) {
      float bm = xormax16(fmaxf(pv[0][i], pv[1][i]));
      const float mnew  = fmaxf(mi[i], bm);
      const float alpha = __expf(mi[i] - mnew);
      const float p0 = __expf(pv[0][i] - mnew);
      const float p1 = __expf(pv[1][i] - mnew);
      li[i] = li[i] * alpha + xorsum16(p0 + p1);
      mi[i] = mnew;
      pv[0][i] = p0; pv[1][i] = p1;
#pragma unroll
      for (int t = 0; t < 8; ++t) o[t][i] *= alpha;
    }

    // P: C-layout -> A-layout via per-wave LDS scratch
#pragma unroll
    for (int nt = 0; nt < 2; ++nt)
#pragma unroll
      for (int i = 0; i < 8; ++i)
        sP[w][(i + 8 * half) * 32 + nt * 16 + l15] = f2bf(pv[nt][i]);
    __syncthreads();
    const int offP = half * 8;
    v16bf pf = cat16(*(const v8bf*)&sP[w][l15 * 32 + offP],
                     *(const v8bf*)&sP[w][l15 * 32 + offP + 16]);

    // ctx: O(16x128) += P(16x32) * V(32x128)
    const int kOff = half * 8;
#pragma unroll
    for (int nt2 = 0; nt2 < 8; ++nt2) {
      const int d = nt2 * 16 + l15;
      v16bf vfr = cat16(*(const v8bf*)&sVt[d * 32 + kOff],
                        *(const v8bf*)&sVt[d * 32 + kOff + 16]);
      o[nt2] = wmma_bf16(pf, vfr, o[nt2]);
    }
  }

  // normalize + emit ctx bf16 at [b*S+s][h*128+d]
#pragma unroll
  for (int i = 0; i < 8; ++i) {
    const float inv = 1.0f / li[i];
    const int s = qBase + i + 8 * half;
    const size_t rowOff = ((size_t)b * Sc + s) * Dc + h * DKc;
#pragma unroll
    for (int nt2 = 0; nt2 < 8; ++nt2)
      ctx[rowOff + nt2 * 16 + l15] = f2bf(o[nt2][i] * inv);
  }
}

// ---------------- kernel C: output projection ----------------

__global__ __launch_bounds__(256) void out_gemm_kernel(
    const unsigned short* __restrict__ Ctx, const unsigned short* __restrict__ WoT,
    const float* __restrict__ bo, float* __restrict__ out) {
  __shared__ __attribute__((aligned(16))) unsigned short sA[2][128 * 64];
  __shared__ __attribute__((aligned(16))) unsigned short sB[2][128 * 64];
  v8f acc[2][4];
#pragma unroll
  for (int mt = 0; mt < 2; ++mt)
#pragma unroll
    for (int nt = 0; nt < 4; ++nt) acc[mt][nt] = v8f{};

  const int m0 = blockIdx.y * 128;
  const int n0 = blockIdx.x * 128;
  gemm_tile(Ctx, WoT, m0, n0, Dc, sA, sB, acc);

  const int lane  = threadIdx.x & 31;
  const int w     = threadIdx.x >> 5;
  const int waveM = w & 3, waveN = w >> 2;
  const int half  = lane >> 4, l15 = lane & 15;
#pragma unroll
  for (int nt = 0; nt < 4; ++nt) {
    const int colN = n0 + waveN * 64 + nt * 16 + l15;
    const float bias = bo[colN];
#pragma unroll
    for (int mt = 0; mt < 2; ++mt) {
#pragma unroll
      for (int i = 0; i < 8; ++i) {
        const int rowM = m0 + waveM * 32 + mt * 16 + i + 8 * half;
        out[(size_t)rowM * Dc + colN] = acc[mt][nt][i] + bias;
      }
    }
  }
}

// ---------------- launch ----------------

extern "C" void kernel_launch(void* const* d_in, const int* in_sizes, int n_in,
                              void* d_out, int out_size, void* d_ws, size_t ws_size,
                              hipStream_t stream) {
  (void)in_sizes; (void)n_in; (void)out_size; (void)ws_size;
  const float*         hidden  = (const float*)d_in[0];
  const unsigned char* seqmask = (const unsigned char*)d_in[1];   // bool mask
  const float*         w_qkv   = (const float*)d_in[2];
  const float*         b_qkv   = (const float*)d_in[3];
  const float*         w_o     = (const float*)d_in[4];
  const float*         b_o     = (const float*)d_in[5];
  float*               out     = (float*)d_out;

  char* ws = (char*)d_ws;
  // workspace layout (bytes): total ~100.7 MB
  unsigned short* Xbf   = (unsigned short*)(ws + 0);           // 16,777,216 (aliased by Ctx)
  unsigned short* WqkvT = (unsigned short*)(ws + 16777216);    // 25,165,824
  unsigned short* WoT   = (unsigned short*)(ws + 41943040);    //  8,388,608
  unsigned short* Qb    = (unsigned short*)(ws + 50331648);    // 16,777,216
  unsigned short* Kb    = (unsigned short*)(ws + 67108864);    // 16,777,216
  unsigned short* Vb    = (unsigned short*)(ws + 83886080);    // 16,777,216
  unsigned short* Ctx   = Xbf;                                 // dead after QKV GEMM

  cvt_f32_bf16_kernel<<<(Mc * Dc) / (256 * 4), 256, 0, stream>>>(hidden, Xbf, Mc * Dc);
  transpose_f32_bf16_kernel<<<dim3(N3c / 32, Dc / 32), 256, 0, stream>>>(w_qkv, WqkvT, Dc, N3c);
  transpose_f32_bf16_kernel<<<dim3(Dc / 32, Dc / 32), 256, 0, stream>>>(w_o, WoT, Dc, Dc);
  qkv_gemm_kernel<<<dim3(N3c / 128, Mc / 128), 256, 0, stream>>>(Xbf, WqkvT, b_qkv, Qb, Kb, Vb);
  attn_kernel<<<dim3(Sc / 64, Hc, Bc), 128, 0, stream>>>(Qb, Kb, Vb, seqmask, Ctx);
  out_gemm_kernel<<<dim3(Dc / 128, Mc / 128), 256, 0, stream>>>(Ctx, WoT, b_o, out);
}
